// CausalSelfAttention_28286654612155
// MI455X (gfx1250) — matrix operand order repaired and checked by reading, not compiled, not run on hardware
//
#include <hip/hip_runtime.h>
#include <hip/hip_bf16.h>

#define BATCH 4
#define SEQ   4096
#define DM    1024
#define HD    64
#define BT    (BATCH * SEQ)   // 16384

typedef __attribute__((ext_vector_type(16))) __bf16 bf16x16;
typedef __attribute__((ext_vector_type(8)))  float  f32x8;
typedef __attribute__((ext_vector_type(4)))  unsigned u32x4;
typedef __attribute__((ext_vector_type(8)))  int      i32x8;
typedef __attribute__((ext_vector_type(4)))  int      i32x4;

union Frag16 {           // 16 bf16 = 32 bytes = one WMMA A/B operand (8 VGPRs)
    bf16x16 v;
    __bf16  h[16];
    uint4   q[2];
};

__device__ __forceinline__ unsigned lds_addr(const void* p) {
    // LDS aperture: generic addr[31:0] == LDS byte offset
    return (unsigned)(uintptr_t)p;
}

__device__ __forceinline__ void async_b128(unsigned ldsdst, const void* gsrc) {
    unsigned long long s64 = (unsigned long long)(uintptr_t)gsrc;
    asm volatile("global_load_async_to_lds_b128 %0, %1, off"
                 :: "v"(ldsdst), "v"(s64) : "memory");
}

// ---------------------------------------------------------------------------
// TDM: 2D tile load Global -> LDS with LDS row padding (bank-conflict-free).
// data_size = 4 bytes. pad: after 2^(pi+1) dwords insert (pa+1) dwords.
// ---------------------------------------------------------------------------
__device__ __forceinline__ void tdm_load_2d(
    unsigned lds, const void* gaddr,
    unsigned tile_d0, unsigned tile_d1,          // tile dims (dwords / rows)
    unsigned tensor_d0, unsigned tensor_d1,      // tensor dims (dwords / rows)
    unsigned stride0,                            // row stride (dwords)
    unsigned pi, unsigned pa) {                  // pad interval / amount codes
    unsigned long long ga = (unsigned long long)(uintptr_t)gaddr;
    u32x4 g0;
    g0[0] = 1u;                                            // count=1, user mode
    g0[1] = lds;                                           // lds_addr
    g0[2] = (unsigned)(ga & 0xFFFFFFFFu);                  // global_addr[31:0]
    g0[3] = (unsigned)((ga >> 32) & 0x1FFFFFFu) | (2u << 30); // addr[56:32], type=2
    i32x8 g1;
    g1[0] = (int)((2u << 16) | (1u << 20) | (pi << 22) | (pa << 25)); // dsz=4B, pad
    g1[1] = (int)((tensor_d0 & 0xFFFFu) << 16);            // tensor_dim0 lo16
    g1[2] = (int)(((tensor_d0 >> 16) & 0xFFFFu) | ((tensor_d1 & 0xFFFFu) << 16));
    g1[3] = (int)(((tensor_d1 >> 16) & 0xFFFFu) | ((tile_d0 & 0xFFFFu) << 16));
    g1[4] = (int)(tile_d1 & 0xFFFFu);                      // tile_dim1, tile_dim2=0
    g1[5] = (int)stride0;                                  // tensor_dim0_stride lo32
    g1[6] = 0;
    g1[7] = 0;
    i32x4 z = {};
#if __clang_major__ >= 23
    i32x8 z8 = {};
    __builtin_amdgcn_tensor_load_to_lds(g0, g1, z, z, z8, 0);
#else
    __builtin_amdgcn_tensor_load_to_lds(g0, g1, z, z, 0);
#endif
}

// ---------------------------------------------------------------------------
// Kernel 1: transpose + convert weights: Wt[w][n][k] = bf16(W_w[k][n])
// ---------------------------------------------------------------------------
__global__ __launch_bounds__(256) void prep_weights(
    const float* __restrict__ Wq, const float* __restrict__ Wk,
    const float* __restrict__ Wv, __bf16* __restrict__ Wt) {
    int idx = blockIdx.x * 256 + threadIdx.x;      // 3 * 64 * 1024 total
    int w = idx >> 16;
    int r = idx & 0xFFFF;
    int n = r >> 10;
    int kd = r & 1023;
    const float* W = (w == 0) ? Wq : (w == 1) ? Wk : Wv;
    Wt[(size_t)w * (HD * DM) + (size_t)n * DM + kd] = (__bf16)W[(size_t)kd * HD + n];
}

// ---------------------------------------------------------------------------
// Kernel 2: fused QKV projection. grid = BT/16, block = 128 (4 waves).
// x tile (16x32 f32) async-staged to LDS, double-buffered; 3 WMMAs/k-step.
// q pre-scaled by 1/sqrt(DM) = 1/32.
// ---------------------------------------------------------------------------
__global__ __launch_bounds__(128) void qkv_proj(
    const float* __restrict__ x, const __bf16* __restrict__ Wt,
    const float* __restrict__ bq, const float* __restrict__ bk,
    const float* __restrict__ bv, __bf16* __restrict__ qkv) {

    __shared__ __attribute__((aligned(16))) float Xb[2][16][36];   // padded rows

    const int tid  = threadIdx.x;
    const int lane = tid & 31;
    const int wave = tid >> 5;
    const int m    = lane & 15;
    const int kh   = lane >> 4;
    const int col  = wave * 16 + m;

    const size_t row0 = (size_t)blockIdx.x * 16;

    const int srow = tid >> 3;
    const int sseg = tid & 7;
    const float* xsrc = x + (row0 + srow) * DM + sseg * 4;
    unsigned xdst[2];
    xdst[0] = lds_addr(&Xb[0][srow][sseg * 4]);
    xdst[1] = lds_addr(&Xb[1][srow][sseg * 4]);

    f32x8 acc0 = {}, acc1 = {}, acc2 = {};

    async_b128(xdst[0], xsrc);                       // prologue stage

    for (int kk = 0; kk < DM; kk += 32) {
        const int cur = (kk >> 5) & 1;
        if (kk + 32 < DM) {
            async_b128(xdst[cur ^ 1], xsrc + kk + 32);    // prefetch next tile
            asm volatile("s_wait_asynccnt 0x1" ::: "memory");
        } else {
            asm volatile("s_wait_asynccnt 0x0" ::: "memory");
        }
        __syncthreads();

        Frag16 a;
        const float* xr = &Xb[cur][m][0];
        float4 fa = *(const float4*)(xr + kh * 16);
        float4 fb = *(const float4*)(xr + kh * 16 + 4);
        float4 fc = *(const float4*)(xr + kh * 16 + 8);
        float4 fd = *(const float4*)(xr + kh * 16 + 12);
        a.h[0]  = (__bf16)fa.x; a.h[1]  = (__bf16)fa.y; a.h[2]  = (__bf16)fa.z; a.h[3]  = (__bf16)fa.w;
        a.h[4]  = (__bf16)fb.x; a.h[5]  = (__bf16)fb.y; a.h[6]  = (__bf16)fb.z; a.h[7]  = (__bf16)fb.w;
        a.h[8]  = (__bf16)fc.x; a.h[9]  = (__bf16)fc.y; a.h[10] = (__bf16)fc.z; a.h[11] = (__bf16)fc.w;
        a.h[12] = (__bf16)fd.x; a.h[13] = (__bf16)fd.y; a.h[14] = (__bf16)fd.z; a.h[15] = (__bf16)fd.w;

        const size_t wko = (size_t)col * DM + kk + kh * 16;
        const uint4* w0 = (const uint4*)(Wt + wko);
        const uint4* w1 = (const uint4*)(Wt + (size_t)HD * DM + wko);
        const uint4* w2 = (const uint4*)(Wt + (size_t)2 * HD * DM + wko);
        Frag16 b0, b1, b2;
        b0.q[0] = w0[0]; b0.q[1] = w0[1];
        b1.q[0] = w1[0]; b1.q[1] = w1[1];
        b2.q[0] = w2[0]; b2.q[1] = w2[1];

        acc0 = __builtin_amdgcn_wmma_f32_16x16x32_bf16(false, a.v, false, b0.v, (short)0, acc0, false, false);
        acc1 = __builtin_amdgcn_wmma_f32_16x16x32_bf16(false, a.v, false, b1.v, (short)0, acc1, false, false);
        acc2 = __builtin_amdgcn_wmma_f32_16x16x32_bf16(false, a.v, false, b2.v, (short)0, acc2, false, false);

        __syncthreads();
    }

    const float bbq = bq[col];
    const float bbk = bk[col];
    const float bbv = bv[col];
    __bf16* dq = qkv;
    __bf16* dk = qkv + (size_t)BT * HD;
    __bf16* dv = qkv + (size_t)2 * BT * HD;
#pragma unroll
    for (int r = 0; r < 8; ++r) {
        const size_t row = row0 + r + 8 * kh;
        dq[row * HD + col] = (__bf16)((acc0[r] + bbq) * (1.0f / 32.0f));
        dk[row * HD + col] = (__bf16)(acc1[r] + bbk);
        dv[row * HD + col] = (__bf16)(acc2[r] + bbv);
    }
}

// ---------------------------------------------------------------------------
// Kernel 2b: transpose v [BT][64] -> vT [B][64][SEQ] (LDS-tiled, coalesced)
// ---------------------------------------------------------------------------
__global__ __launch_bounds__(256) void transpose_v(
    const __bf16* __restrict__ v, __bf16* __restrict__ vT) {
    __shared__ __attribute__((aligned(16))) __bf16 tile[64][72];
    const int b  = blockIdx.y;
    const int t0 = blockIdx.x * 64;
    const int tid = threadIdx.x;
    const int r  = tid >> 2;          // 0..63
    const int sg = tid & 3;           // 16 elements each

    Frag16 tmp;
    const uint4* src = (const uint4*)(v + ((size_t)b * SEQ + t0 + r) * HD + sg * 16);
    tmp.q[0] = src[0];
    tmp.q[1] = src[1];
#pragma unroll
    for (int e = 0; e < 16; ++e) tile[sg * 16 + e][r] = tmp.h[e];   // tile[d][t]
    __syncthreads();

    Frag16 o;
#pragma unroll
    for (int e = 0; e < 16; ++e) o.h[e] = tile[r][sg * 16 + e];
    uint4* dst = (uint4*)(vT + ((size_t)b * HD + r) * SEQ + t0 + sg * 16);
    dst[0] = o.q[0];
    dst[1] = o.q[1];
}

// ---------------------------------------------------------------------------
// Kernel 3: causal flash attention. grid = (SEQ/64, BATCH), block = 128.
// K and V^T tiles are pulled by the Tensor Data Mover into padded LDS rows,
// double-buffered (TENSORcnt pipeline). All fragment reads: 2x ds_load_b128.
// ---------------------------------------------------------------------------
__global__ __launch_bounds__(128) void attn_flash(
    const __bf16* __restrict__ q, const __bf16* __restrict__ k,
    const __bf16* __restrict__ vT, float* __restrict__ out) {

    __shared__ __attribute__((aligned(16))) __bf16 Kt[2][32][72];   // key-major, TDM-padded
    __shared__ __attribute__((aligned(16))) __bf16 Vt[2][64][40];   // d-major,  TDM-padded
    __shared__ __attribute__((aligned(16))) __bf16 Pbuf[4][16][40]; // per-wave P tile

    const int tid  = threadIdx.x;
    const int lane = tid & 31;
    const int wave = tid >> 5;
    const int n16  = lane & 15;
    const int kh   = lane >> 4;
    const int b    = blockIdx.y;
    const int q0b  = blockIdx.x * 64;
    const int q0   = q0b + wave * 16;
    const size_t base = (size_t)b * SEQ;

    // ---- Q fragments (16 rows x 64 dims as two 16x32 A frags) ----
    Frag16 qa[2];
    {
        const uint4* qrow = (const uint4*)(q + (base + q0 + n16) * HD);
        qa[0].q[0] = qrow[2 * kh];
        qa[0].q[1] = qrow[2 * kh + 1];
        qa[1].q[0] = qrow[4 + 2 * kh];
        qa[1].q[1] = qrow[5 + 2 * kh];
    }

    float M[8], L[8];
    f32x8 o[4];
#pragma unroll
    for (int r = 0; r < 8; ++r) { M[r] = -INFINITY; L[r] = 0.0f; }
#pragma unroll
    for (int t = 0; t < 4; ++t) o[t] = (f32x8){};

    const int nj = ((q0b + 63) >> 5) + 1;            // key tiles covering keys <= q0b+63

    // TDM tile issue: K tile = 32 rows x 32 dw (pad 32dw->+4dw => 72-elem rows)
    //                 V tile = 64 rows x 16 dw (pad 16dw->+4dw => 40-elem rows)
    const __bf16* vTb = vT + (size_t)b * HD * SEQ;
    if (wave == 0) {
        tdm_load_2d(lds_addr(&Kt[0][0][0]), k + (base + 0) * HD,
                    32, 32, 32, SEQ, 32, /*pi=*/4, /*pa=*/3);
        tdm_load_2d(lds_addr(&Vt[0][0][0]), vTb + 0,
                    16, 64, SEQ / 2, HD, SEQ / 2, /*pi=*/3, /*pa=*/3);
    }

    for (int j = 0; j < nj; ++j) {
        const int cur = j & 1;
        const int kbase = j * 32;

        if (wave == 0) {
            if (j + 1 < nj) {
                const int kn = (j + 1) * 32;
                tdm_load_2d(lds_addr(&Kt[cur ^ 1][0][0]), k + (base + kn) * HD,
                            32, 32, 32, SEQ, 32, 4, 3);
                tdm_load_2d(lds_addr(&Vt[cur ^ 1][0][0]), vTb + kn,
                            16, 64, SEQ / 2, HD, SEQ / 2, 3, 3);
                __builtin_amdgcn_s_wait_tensorcnt(2);   // pair j complete
            } else {
                __builtin_amdgcn_s_wait_tensorcnt(0);
            }
        }
        __syncthreads();

        // ---- S = Q @ K^T : two 16x16 n-tiles, head dim 64 in two steps ----
        f32x8 s[2];
#pragma unroll
        for (int t = 0; t < 2; ++t) {
            s[t] = (f32x8){};
#pragma unroll
            for (int d = 0; d < 2; ++d) {
                Frag16 kb;                           // B 32x16: n=key, K=head dim
                const uint4* kr = (const uint4*)&Kt[cur][t * 16 + n16][d * 32 + kh * 16];
                kb.q[0] = kr[0];
                kb.q[1] = kr[1];
                s[t] = __builtin_amdgcn_wmma_f32_16x16x32_bf16(
                    false, qa[d].v, false, kb.v, (short)0, s[t], false, false);
            }
        }

        // ---- causal mask + online softmax (C layout: m = r + 8*kh, n = n16) ----
#pragma unroll
        for (int r = 0; r < 8; ++r) {
            const int qrow_g = q0 + r + 8 * kh;
            float s0 = (kbase + n16      <= qrow_g) ? s[0][r] : -INFINITY;
            float s1 = (kbase + 16 + n16 <= qrow_g) ? s[1][r] : -INFINITY;

            float mx = fmaxf(s0, s1);
#pragma unroll
            for (int off = 1; off < 16; off <<= 1)
                mx = fmaxf(mx, __shfl_xor(mx, off, 16));

            const float newM = fmaxf(M[r], mx);
            const float corr = __expf(M[r] - newM);
            M[r] = newM;

            const float p0 = __expf(s0 - newM);
            const float p1 = __expf(s1 - newM);
            float sum = p0 + p1;
#pragma unroll
            for (int off = 1; off < 16; off <<= 1)
                sum += __shfl_xor(sum, off, 16);

            L[r] = L[r] * corr + sum;
#pragma unroll
            for (int t = 0; t < 4; ++t) o[t][r] *= corr;

            Pbuf[wave][r + 8 * kh][n16]      = (__bf16)p0;
            Pbuf[wave][r + 8 * kh][16 + n16] = (__bf16)p1;
        }

        // ---- read P back as a 16x32 A fragment (in-wave LDS transpose) ----
        Frag16 pa;
        {
            const uint4* prow = (const uint4*)&Pbuf[wave][n16][0];
            pa.q[0] = prow[2 * kh];
            pa.q[1] = prow[2 * kh + 1];
        }

        // ---- O += P @ V : four 16x16 d-tiles from transposed V ----
#pragma unroll
        for (int t = 0; t < 4; ++t) {
            Frag16 vb;                               // B 32x16: n=dim, K=key
            const uint4* vr = (const uint4*)&Vt[cur][t * 16 + n16][kh * 16];
            vb.q[0] = vr[0];
            vb.q[1] = vr[1];
            o[t] = __builtin_amdgcn_wmma_f32_16x16x32_bf16(
                false, pa.v, false, vb.v, (short)0, o[t], false, false);
        }

        __syncthreads();                              // buffer reused after this
    }

    // ---- normalize and write out (f32) ----
#pragma unroll
    for (int r = 0; r < 8; ++r) {
        const float inv = 1.0f / L[r];
        const size_t row = base + q0 + r + 8 * kh;
#pragma unroll
        for (int t = 0; t < 4; ++t)
            out[row * HD + t * 16 + n16] = o[t][r] * inv;
    }
}

// ---------------------------------------------------------------------------
extern "C" void kernel_launch(void* const* d_in, const int* in_sizes, int n_in,
                              void* d_out, int out_size, void* d_ws, size_t ws_size,
                              hipStream_t stream) {
    const float* x  = (const float*)d_in[0];
    const float* Wq = (const float*)d_in[1];
    const float* bq = (const float*)d_in[2];
    const float* Wk = (const float*)d_in[3];
    const float* bk = (const float*)d_in[4];
    const float* Wv = (const float*)d_in[5];
    const float* bv = (const float*)d_in[6];

    char* ws = (char*)d_ws;
    __bf16* Wt  = (__bf16*)ws;                                   // 384 KB
    __bf16* qkv = (__bf16*)(ws + (size_t)3 * HD * DM * 2);       // 6 MB
    __bf16* qb  = qkv;
    __bf16* kb  = qkv + (size_t)BT * HD;
    __bf16* vb  = qkv + (size_t)2 * BT * HD;
    __bf16* vT  = qkv + (size_t)3 * BT * HD;                     // 2 MB  [B][64][SEQ]

    prep_weights<<<(3 * HD * DM) / 256, 256, 0, stream>>>(Wq, Wk, Wv, Wt);
    qkv_proj<<<BT / 16, 128, 0, stream>>>(x, Wt, bq, bk, bv, qkv);
    transpose_v<<<dim3(SEQ / 64, BATCH), 256, 0, stream>>>(vb, vT);
    attn_flash<<<dim3(SEQ / 64, BATCH), 128, 0, stream>>>(qb, kb, vT, (float*)d_out);
}
